// Int8GPT2MLP_6511170421642
// MI455X (gfx1250) — compile-verified, matrix operand order checked by reading, not run on hardware
//
#include <hip/hip_runtime.h>
#include <hip/hip_bf16.h>

typedef __attribute__((ext_vector_type(8))) int v8i;
typedef __attribute__((ext_vector_type(4))) int v4i_t;
typedef __attribute__((address_space(1))) v4i_t gv4i;
typedef __attribute__((address_space(3))) v4i_t lv4i;

#define TILE_M 128
#define TILE_N 128
#define TILE_K 64

// ---- async 16B global -> LDS copy (CDNA5 ASYNCcnt path) ----
// flat->AS1 keeps the full 64-bit address; flat->AS3 keeps the low 32 bits
// (LDS aperture: LDS offset is the low 32 bits).
__device__ inline void async_copy_b128(void* lds_dst, const void* gsrc) {
#if defined(__has_builtin) && __has_builtin(__builtin_amdgcn_global_load_async_to_lds_b128)
  __builtin_amdgcn_global_load_async_to_lds_b128(
      (gv4i*)(unsigned long long)(uintptr_t)gsrc,
      (lv4i*)(unsigned int)(uintptr_t)lds_dst,
      0, 0);
#else
  asm volatile("global_load_async_to_lds_b128 %0, %1, off"
               :: "v"((lv4i*)(unsigned int)(uintptr_t)lds_dst),
                  "v"((gv4i*)(unsigned long long)(uintptr_t)gsrc)
               : "memory");
#endif
}

__device__ inline void wait_async_zero() {
#if defined(__has_builtin) && __has_builtin(__builtin_amdgcn_s_wait_asynccnt)
  __builtin_amdgcn_s_wait_asynccnt(0);
#else
  asm volatile("s_wait_asynccnt 0" ::: "memory");
#endif
}

// ---------------------------------------------------------------------------
// Pack int8-valued int32 array into contiguous int8 bytes (row-major kept).
// ---------------------------------------------------------------------------
__global__ __launch_bounds__(256) void pack_i32_to_i8(const int* __restrict__ src,
                                                      signed char* __restrict__ dst,
                                                      int n4) {
  int i = blockIdx.x * blockDim.x + threadIdx.x;
  if (i >= n4) return;
  int j = i << 2;
  unsigned b0 = (unsigned)(src[j + 0] & 0xff);
  unsigned b1 = (unsigned)(src[j + 1] & 0xff);
  unsigned b2 = (unsigned)(src[j + 2] & 0xff);
  unsigned b3 = (unsigned)(src[j + 3] & 0xff);
  ((unsigned*)dst)[i] = b0 | (b1 << 8) | (b2 << 16) | (b3 << 24);
}

// ---------------------------------------------------------------------------
// Transpose-pack weight: w[K][N] int32 (int8 values) -> wp[N][K] int8 bytes.
// ---------------------------------------------------------------------------
__global__ __launch_bounds__(256) void pack_w_transpose(const int* __restrict__ w,
                                                        signed char* __restrict__ wp,
                                                        int K, int N) {
  __shared__ __attribute__((aligned(16))) signed char tile[64 * 80]; // [n][k], stride 80

  const int nbase = blockIdx.x * 64;
  const int kbase = blockIdx.y * 64;
  const int t = threadIdx.x;

  #pragma unroll
  for (int e = t; e < 64 * 64; e += 256) {
    int k = e >> 6;
    int n = e & 63;
    int v = w[(size_t)(kbase + k) * N + (nbase + n)];
    tile[n * 80 + k] = (signed char)v;
  }
  __syncthreads();

  int n = t >> 2;
  int chunk = (t & 3) * 16;
  uint4 v = *(const uint4*)(&tile[n * 80 + chunk]);
  *(uint4*)(&wp[(size_t)(nbase + n) * K + kbase + chunk]) = v;
}

// ---------------------------------------------------------------------------
// Int8 WMMA GEMM:  C[M][N] = A[M][K] * B^T  where B is stored N-major [N][K].
// Block tile 128x128, K-step 64, 8 waves (2x4), each wave: 4x2 16x16 tiles.
// Global->LDS staging uses async-to-LDS copies, double buffered.
// EPI==0: requant epilogue -> int8 out, staged through LDS for b128 stores.
// EPI==1: dequant epilogue -> fp32 out + bias.
// ---------------------------------------------------------------------------
template <int EPI>
__global__ __launch_bounds__(256) void wmma_i8_gemm(
    const signed char* __restrict__ A,   // [M][K] int8
    const signed char* __restrict__ B,   // [N][K] int8 (column-major of [K][N])
    int M, int N, int K,
    signed char* __restrict__ outq, const int* __restrict__ bias_i8,
    const float* __restrict__ alpha_p, const float* __restrict__ beta_p,
    float* __restrict__ outf, const float* __restrict__ bias_f) {
  __shared__ __attribute__((aligned(16))) signed char As[2][TILE_M * TILE_K];
  __shared__ __attribute__((aligned(16))) signed char Bs[2][TILE_N * TILE_K];

  const int tid   = threadIdx.x;
  const int lane  = tid & 31;
  const int wave  = tid >> 5;
  const int wm    = wave & 1;   // 0..1  -> M offset 0/64
  const int wn    = wave >> 1;  // 0..3  -> N offset 0/32/64/96
  const int lrow  = lane & 15;  // row (A) / col (B) within 16x16 tile
  const int lhalf = lane >> 4;  // K-half selector per ISA 8-bit layout

  const int bm = blockIdx.x;
  const int bn = blockIdx.y;

  const size_t a_base = (size_t)bm * TILE_M * (size_t)K;
  const size_t b_base = (size_t)bn * TILE_N * (size_t)K;

  const float alpha = alpha_p[0];
  const float beta  = (EPI == 0) ? beta_p[0] : 0.0f;

  v8i acc[4][2];
  const v8i vzero = {0, 0, 0, 0, 0, 0, 0, 0};
  #pragma unroll
  for (int mt = 0; mt < 4; ++mt)
    #pragma unroll
    for (int nt = 0; nt < 2; ++nt) acc[mt][nt] = vzero;

  // 128 rows x 64 bytes per tile, 256 threads, 32 bytes per thread per tile
  const int crow = tid >> 1;
  const int coff = (tid & 1) * 32;

  auto copy_stage_async = [&](int kt, int buf) {
    const signed char* as = A + a_base + (size_t)crow * K + (size_t)kt * TILE_K + coff;
    async_copy_b128(&As[buf][crow * TILE_K + coff],      as);
    async_copy_b128(&As[buf][crow * TILE_K + coff + 16], as + 16);
    const signed char* bs = B + b_base + (size_t)crow * K + (size_t)kt * TILE_K + coff;
    async_copy_b128(&Bs[buf][crow * TILE_K + coff],      bs);
    async_copy_b128(&Bs[buf][crow * TILE_K + coff + 16], bs + 16);
  };

  // 8-bit operand fragment per ISA layout:
  //   lanes 0-15 : K = {0-7, 16-23, 32-39, 48-55} of row (lane&15)
  //   lanes 16-31: same rows, K shifted +8
  auto load_frag = [&](const signed char* p) -> v8i {
    int2 d0 = *(const int2*)(p + 0);
    int2 d1 = *(const int2*)(p + 16);
    int2 d2 = *(const int2*)(p + 32);
    int2 d3 = *(const int2*)(p + 48);
    v8i f;
    f[0] = d0.x; f[1] = d0.y;
    f[2] = d1.x; f[3] = d1.y;
    f[4] = d2.x; f[5] = d2.y;
    f[6] = d3.x; f[7] = d3.y;
    return f;
  };

  const int KT = K / TILE_K;
  copy_stage_async(0, 0);
  wait_async_zero();
  __syncthreads();

  int buf = 0;
  for (int kt = 0; kt < KT; ++kt) {
    // kick off next stage; it runs concurrently with the WMMAs below
    if (kt + 1 < KT) copy_stage_async(kt + 1, buf ^ 1);

    v8i afrag[4];
    #pragma unroll
    for (int mt = 0; mt < 4; ++mt) {
      int row = wm * 64 + mt * 16 + lrow;
      afrag[mt] = load_frag(&As[buf][row * TILE_K + lhalf * 8]);
    }
    #pragma unroll
    for (int nt = 0; nt < 2; ++nt) {
      int col = wn * 32 + nt * 16 + lrow;
      v8i bfrag = load_frag(&Bs[buf][col * TILE_K + lhalf * 8]);
      #pragma unroll
      for (int mt = 0; mt < 4; ++mt) {
        acc[mt][nt] = __builtin_amdgcn_wmma_i32_16x16x64_iu8(
            /*sgn_a=*/true, afrag[mt], /*sgn_b=*/true, bfrag,
            acc[mt][nt], /*reuse_a=*/false, /*reuse_b=*/false);
      }
    }
    wait_async_zero();   // our async writes for kt+1 have landed in LDS
    __syncthreads();     // everyone done reading buf / writing buf^1
    buf ^= 1;
  }

  // ---- epilogue ----
  // C/D layout: vgpr r, lanes 0-15 -> (M=r, N=lane); lanes 16-31 -> (M=r+8, N=lane-16)
  if (EPI == 0) {
    // Stage the 128x128 int8 tile in LDS (reuse As: 2*8KB = exactly 128*128B),
    // then write back with coalesced 16B stores.
    signed char* stage = &As[0][0];
    #pragma unroll
    for (int mt = 0; mt < 4; ++mt) {
      #pragma unroll
      for (int nt = 0; nt < 2; ++nt) {
        int ln = wn * 32 + nt * 16 + lrow;
        int gn = bn * TILE_N + ln;
        float bterm = (float)bias_i8[gn] * beta;
        #pragma unroll
        for (int r = 0; r < 8; ++r) {
          int lm = wm * 64 + mt * 16 + r + lhalf * 8;
          float h = (float)acc[mt][nt][r] * alpha + bterm;
          h = fmaxf(h, 0.0f);
          h = rintf(h);
          h = fminf(h, 127.0f);
          stage[lm * TILE_N + ln] = (signed char)(int)h;
        }
      }
    }
    __syncthreads();
    #pragma unroll
    for (int e = tid; e < TILE_M * (TILE_N / 16); e += 256) {
      int row = e >> 3;            // 0..127
      int chunk = (e & 7) * 16;    // 0..112
      uint4 v = *(const uint4*)(&stage[row * TILE_N + chunk]);
      *(uint4*)(&outq[(size_t)(bm * TILE_M + row) * N + bn * TILE_N + chunk]) = v;
    }
  } else {
    #pragma unroll
    for (int mt = 0; mt < 4; ++mt) {
      #pragma unroll
      for (int nt = 0; nt < 2; ++nt) {
        int gn = bn * TILE_N + wn * 32 + nt * 16 + lrow;
        float bterm = bias_f[gn];
        #pragma unroll
        for (int r = 0; r < 8; ++r) {
          int gm = bm * TILE_M + wm * 64 + mt * 16 + r + lhalf * 8;
          outf[(size_t)gm * N + gn] = (float)acc[mt][nt][r] * alpha + bterm;
        }
      }
    }
  }
}

// ---------------------------------------------------------------------------
// Launch: pack prepass + two WMMA GEMMs.
// Workspace layout (88 MB):
//   a_pack     [M*H]  int8   (16 MB)
//   wfc_pack   [I*H]  int8   ( 4 MB)  N-major, K=H
//   wproj_pack [H*I]  int8   ( 4 MB)  N-major, K=I
//   hq         [M*I]  int8   (64 MB)  requantized intermediate
// ---------------------------------------------------------------------------
extern "C" void kernel_launch(void* const* d_in, const int* in_sizes, int n_in,
                              void* d_out, int out_size, void* d_ws, size_t ws_size,
                              hipStream_t stream) {
  const int*   hidden     = (const int*)d_in[0];
  const int*   w_fc       = (const int*)d_in[1];
  const int*   b_fc       = (const int*)d_in[2];
  const float* alpha_fc   = (const float*)d_in[3];
  const float* beta_fc    = (const float*)d_in[4];
  const int*   w_proj     = (const int*)d_in[5];
  const float* b_proj     = (const float*)d_in[6];
  const float* alpha_proj = (const float*)d_in[7];
  float*       out        = (float*)d_out;

  const int B = 8, S = 2048, H = 1024, I = 4096;
  const int M = B * S; // 16384

  char* ws = (char*)d_ws;
  signed char* a_pack     = (signed char*)ws;
  signed char* wfc_pack   = a_pack + (size_t)M * H;
  signed char* wproj_pack = wfc_pack + (size_t)I * H;
  signed char* hq         = wproj_pack + (size_t)H * I;

  {
    int n4 = (M * H) / 4;
    pack_i32_to_i8<<<(n4 + 255) / 256, 256, 0, stream>>>(hidden, a_pack, n4);
  }
  {
    dim3 g(I / 64, H / 64);
    pack_w_transpose<<<g, 256, 0, stream>>>(w_fc, wfc_pack, H, I);
  }
  {
    dim3 g(H / 64, I / 64);
    pack_w_transpose<<<g, 256, 0, stream>>>(w_proj, wproj_pack, I, H);
  }
  {
    dim3 grid(M / TILE_M, I / TILE_N);
    wmma_i8_gemm<0><<<grid, 256, 0, stream>>>(a_pack, wfc_pack, M, I, H,
                                              hq, b_fc, alpha_fc, beta_fc,
                                              nullptr, nullptr);
  }
  {
    dim3 grid(M / TILE_M, H / TILE_N);
    wmma_i8_gemm<1><<<grid, 256, 0, stream>>>(hq, wproj_pack, M, H, I,
                                              nullptr, nullptr, alpha_proj, nullptr,
                                              out, b_proj);
  }
}
